// WindowAttention_81982335746153
// MI455X (gfx1250) — matrix operand order repaired and checked by reading, not compiled, not run on hardware
//
#include <hip/hip_runtime.h>
#include <hip/hip_bf16.h>
#include <math.h>

// Swin window attention for MI455X (gfx1250), wave32 + WMMA bf16 path.
// Workspace requirement: 3*24576*64*32 + 2048*64*384 bf16 elems = ~403 MB.

typedef __attribute__((ext_vector_type(4)))  __bf16 v4bf;
typedef __attribute__((ext_vector_type(8)))  __bf16 v8bf;
typedef __attribute__((ext_vector_type(16))) __bf16 v16bf;
typedef __attribute__((ext_vector_type(8)))  float  v8f;

static __device__ __forceinline__ v16bf cat16(v8bf lo, v8bf hi) {
  return __builtin_shufflevector(lo, hi, 0, 1, 2, 3, 4, 5, 6, 7,
                                 8, 9, 10, 11, 12, 13, 14, 15);
}

#define CDIM 384
#define NH   12
#define HD   32
#define NTOK 49
#define NPAD 64
#define NWIN 2048              // 32 batches * 64 windows
#define NWH  (NWIN * NH)       // 24576
#define QKV_COLS 1152

// workspace offsets (bf16 element units)
#define OFF_Q  0
#define OFF_K  (NWH * NPAD * HD)        // 50331648
#define OFF_V  (2 * NWH * NPAD * HD)    // V stored transposed: [wh][hd][tok]
#define OFF_OH (3 * NWH * NPAD * HD)    // head-concat output: [win][tok64][384]

// ---------------------------------------------------------------------------
// Kernel 1: fused QKV projection.  x(window-partitioned) @ qkv_w + qkv_b
// 128 threads (4 waves); block tile = 16 rows x 128 cols; 2 col-tiles/wave.
// ---------------------------------------------------------------------------
__global__ __launch_bounds__(128) void swin_qkv_kernel(
    const float* __restrict__ hidden, const float* __restrict__ qkv_w,
    const float* __restrict__ qkv_b, __bf16* __restrict__ ws)
{
  __shared__ __align__(32) __bf16 sA[16 * 32];    // [tok][k]
  __shared__ __align__(32) __bf16 sBt[128 * 32];  // [col][k] (transposed)

  const int tid  = threadIdx.x;
  const int wave = tid >> 5, lane = tid & 31;
  const int lrow = lane & 15, g = lane >> 4;

  const int win  = blockIdx.x >> 2;
  const int row0 = (blockIdx.x & 3) << 4;
  const int c0   = blockIdx.y << 7;   // 128 cols per block

  const int  b  = win >> 6;
  const int  wy = (win & 63) >> 3;
  const int  wx = win & 7;
  const long hbase = (long)b * (56 * 56);

  // A-tile address for this thread's float4 (one per thread per K-step)
  const int ar  = tid >> 3;            // token row 0..15
  const int acg = (tid & 7) << 2;      // channel group *4
  const int atok = row0 + ar;
  long aoff = -1;
  if (atok < NTOK) {
    int ty = atok / 7, tx = atok % 7;
    long L = (long)(wy * 7 + ty) * 56 + (wx * 7 + tx);
    aoff = (hbase + L) * CDIM + acg;
  }

  v8f acc[2] = {{}, {}};
  for (int kk = 0; kk < CDIM; kk += 32) {
    // stage A: 16x32 f32 -> bf16 (one float4 per thread)
    float4 wv = make_float4(0.f, 0.f, 0.f, 0.f);
    if (aoff >= 0) wv = *(const float4*)&hidden[aoff + kk];
    *(v4bf*)&sA[ar * 32 + acg] =
        (v4bf){(__bf16)wv.x, (__bf16)wv.y, (__bf16)wv.z, (__bf16)wv.w};

    // stage B transposed: sBt[col][k]  (8 float4 per thread)
    for (int i = tid; i < 1024; i += 128) {
      int r = i >> 5, cg = (i & 31) << 2;
      float4 q = *(const float4*)&qkv_w[(long)(kk + r) * QKV_COLS + c0 + cg];
      sBt[(cg + 0) * 32 + r] = (__bf16)q.x;
      sBt[(cg + 1) * 32 + r] = (__bf16)q.y;
      sBt[(cg + 2) * 32 + r] = (__bf16)q.z;
      sBt[(cg + 3) * 32 + r] = (__bf16)q.w;
    }
    __syncthreads();

    // A loaded once, reused for both column tiles
    v8bf alo = *(const v8bf*)&sA[lrow * 32 + g * 8];
    v8bf ahi = *(const v8bf*)&sA[lrow * 32 + 16 + g * 8];
    v16bf a  = cat16(alo, ahi);
#pragma unroll
    for (int t = 0; t < 2; ++t) {
      v16bf bm = cat16(*(const v8bf*)&sBt[(wave * 32 + t * 16 + lrow) * 32 + g * 8],
                       *(const v8bf*)&sBt[(wave * 32 + t * 16 + lrow) * 32 + 16 + g * 8]);
      acc[t] = __builtin_amdgcn_wmma_f32_16x16x32_bf16(false, a, false, bm,
                                                       (short)0, acc[t],
                                                       false, false);
    }
    __syncthreads();
  }

  __bf16* Q  = ws + OFF_Q;
  __bf16* Kt = ws + OFF_K;
  __bf16* Vt = ws + OFF_V;
  const float scale = 0.17677669529663687f;  // hd^-0.5
#pragma unroll
  for (int t = 0; t < 2; ++t) {
    // wave-uniform segment select (16-col tile never straddles Q/K/V bounds)
    const int colbase = __builtin_amdgcn_readfirstlane(c0 + wave * 32 + t * 16);
    const int col  = colbase + lrow;
    const float bias = qkv_b[col];
    const v8f av = acc[t];
    if (colbase < 384) {                     // Q (pre-scaled)
      const int head = col >> 5, d = col & 31;
      __bf16* dst = Q + (long)(win * NH + head) * (NPAD * HD) + d;
#pragma unroll
      for (int i = 0; i < 8; ++i)
        dst[(row0 + i + 8 * g) * HD] = (__bf16)((av[i] + bias) * scale);
    } else if (colbase < 768) {              // K
      const int c2 = col - 384, head = c2 >> 5, d = c2 & 31;
      __bf16* dst = Kt + (long)(win * NH + head) * (NPAD * HD) + d;
#pragma unroll
      for (int i = 0; i < 8; ++i)
        dst[(row0 + i + 8 * g) * HD] = (__bf16)(av[i] + bias);
    } else {                                 // V, stored transposed [hd][tok]
      const int c2 = col - 768, head = c2 >> 5, d = c2 & 31;
      __bf16* dst = Vt + (long)(win * NH + head) * (NPAD * HD) + d * NPAD;
#pragma unroll
      for (int i = 0; i < 8; ++i)
        dst[row0 + i + 8 * g] = (__bf16)(av[i] + bias);
    }
  }
}

// ---------------------------------------------------------------------------
// Kernel 2: per-(window,head) attention: S=Q*K^T, +bias, softmax, O=P*V
// One 128-thread block per (window,head). K/V staged in LDS (shared by all
// waves); Q rows are per-wave private -> loaded directly from global.
// ---------------------------------------------------------------------------
__global__ __launch_bounds__(128) void swin_attn_kernel(
    const float* __restrict__ bias_table, __bf16* __restrict__ ws)
{
  __shared__ __align__(32) __bf16 sK[NPAD * HD];
  __shared__ __align__(32) __bf16 sV[HD * NPAD];   // [d][tok]
  __shared__ __align__(32) float  sS[NPAD * NPAD];
  __shared__ __align__(32) __bf16 sP[NPAD * NPAD];
  __shared__ float sBias[169];

  const int tid  = threadIdx.x;
  const int wave = tid >> 5, lane = tid & 31;
  const int lrow = lane & 15, g = lane >> 4;

  const int wh  = blockIdx.x;
  const int win = wh / NH, head = wh % NH;

  const v8bf* Kg = (const v8bf*)(ws + OFF_K + (long)wh * (NPAD * HD));
  const v8bf* Vg = (const v8bf*)(ws + OFF_V + (long)wh * (NPAD * HD));
  v8bf* sK8 = (v8bf*)sK;
  v8bf* sV8 = (v8bf*)sV;
  for (int i = tid; i < (NPAD * HD) / 8; i += 128) {
    sK8[i] = Kg[i]; sV8[i] = Vg[i];
  }
  for (int i = tid; i < 169; i += 128) sBias[i] = bias_table[i * NH + head];

  // A operand (this wave's 16 Q rows) straight from global, no LDS round-trip
  const __bf16* Qrow =
      ws + OFF_Q + (long)wh * (NPAD * HD) + (wave * 16 + lrow) * HD;
  v8bf qlo = *(const v8bf*)&Qrow[g * 8];
  v8bf qhi = *(const v8bf*)&Qrow[16 + g * 8];
  v16bf aq = cat16(qlo, qhi);
  __syncthreads();

  // S = Q * K^T  (M=64, N=64, K=32): wave w owns rows [16w,16w+16), A reused
#pragma unroll
  for (int nt = 0; nt < 4; ++nt) {
    v16bf bm = cat16(*(const v8bf*)&sK[(nt * 16 + lrow) * HD + g * 8],
                     *(const v8bf*)&sK[(nt * 16 + lrow) * HD + 16 + g * 8]);
    v8f acc = {};
    acc = __builtin_amdgcn_wmma_f32_16x16x32_bf16(false, aq, false, bm,
                                                  (short)0, acc, false, false);
#pragma unroll
    for (int i = 0; i < 8; ++i)
      sS[(wave * 16 + i + 8 * g) * NPAD + nt * 16 + lrow] = acc[i];
  }
  __syncthreads();

  // softmax over 49 valid keys with relative position bias
  if (tid < NPAD) {
    int r = tid;
    if (r < NTOK) {
      int qy = r / 7, qx = r % 7;
      float mx = -1e30f;
      for (int c = 0; c < NTOK; ++c) {
        int ky = c / 7, kx = c % 7;
        float s = sS[r * NPAD + c] + sBias[(qy - ky + 6) * 13 + (qx - kx + 6)];
        sS[r * NPAD + c] = s;
        mx = fmaxf(mx, s);
      }
      float sum = 0.f;
      for (int c = 0; c < NTOK; ++c) {
        float e = __expf(sS[r * NPAD + c] - mx);
        sS[r * NPAD + c] = e;
        sum += e;
      }
      float inv = 1.f / sum;
      for (int c = 0; c < NPAD; ++c)
        sP[r * NPAD + c] = (c < NTOK) ? (__bf16)(sS[r * NPAD + c] * inv)
                                      : (__bf16)0.f;
    } else {
      for (int c = 0; c < NPAD; ++c) sP[r * NPAD + c] = (__bf16)0.f;
    }
  }
  __syncthreads();

  // O = P * V  (M=64, N=32, K=64): wave w owns rows [16w,16w+16);
  // A (P slice) loaded once per K-step, reused for both N tiles.
  __bf16* Oh = ws + OFF_OH;
  v8f acc[2] = {{}, {}};
#pragma unroll
  for (int kk = 0; kk < NPAD; kk += 32) {
    v8bf plo = *(const v8bf*)&sP[(wave * 16 + lrow) * NPAD + kk + g * 8];
    v8bf phi = *(const v8bf*)&sP[(wave * 16 + lrow) * NPAD + kk + 16 + g * 8];
    v16bf a  = cat16(plo, phi);
#pragma unroll
    for (int nt = 0; nt < 2; ++nt) {
      v16bf bm = cat16(*(const v8bf*)&sV[(nt * 16 + lrow) * NPAD + kk + g * 8],
                       *(const v8bf*)&sV[(nt * 16 + lrow) * NPAD + kk + 16 + g * 8]);
      acc[nt] = __builtin_amdgcn_wmma_f32_16x16x32_bf16(false, a, false, bm,
                                                        (short)0, acc[nt],
                                                        false, false);
    }
  }
#pragma unroll
  for (int nt = 0; nt < 2; ++nt) {
#pragma unroll
    for (int i = 0; i < 8; ++i) {
      int m = wave * 16 + i + 8 * g;
      int d = nt * 16 + lrow;
      float v = (m < NTOK) ? acc[nt][i] : 0.f;  // zero pad rows for proj
      Oh[((long)win * NPAD + m) * CDIM + head * HD + d] = (__bf16)v;
    }
  }
}

// ---------------------------------------------------------------------------
// Kernel 3: output projection + window reverse.  Oh @ proj_w + proj_b
// Block tile = 16 rows x 128 cols; 2 col-tiles per wave.
// ---------------------------------------------------------------------------
__global__ __launch_bounds__(128) void swin_proj_kernel(
    const float* __restrict__ proj_w, const float* __restrict__ proj_b,
    const __bf16* __restrict__ ws, float* __restrict__ out)
{
  __shared__ __align__(32) __bf16 sA[16 * 32];    // [tok][k]
  __shared__ __align__(32) __bf16 sBt[128 * 32];  // [col][k] (transposed)

  const int tid  = threadIdx.x;
  const int wave = tid >> 5, lane = tid & 31;
  const int lrow = lane & 15, g = lane >> 4;

  const int win  = blockIdx.x >> 2;
  const int row0 = (blockIdx.x & 3) << 4;
  const int c0   = blockIdx.y << 7;   // 128 cols per block

  const __bf16* Oh = ws + OFF_OH;

  v8f acc[2] = {{}, {}};
  for (int kk = 0; kk < CDIM; kk += 32) {
    // stage A (bf16 copy, 8 elems per chunk; 64 chunks)
    if (tid < 64) {
      int r = tid >> 2, cg = (tid & 3) << 3;
      *(v8bf*)&sA[r * 32 + cg] =
          *(const v8bf*)&Oh[((long)win * NPAD + row0 + r) * CDIM + kk + cg];
    }
    // stage B transposed: sBt[col][k]
    for (int i = tid; i < 1024; i += 128) {
      int r = i >> 5, cg = (i & 31) << 2;
      float4 q = *(const float4*)&proj_w[(long)(kk + r) * CDIM + c0 + cg];
      sBt[(cg + 0) * 32 + r] = (__bf16)q.x;
      sBt[(cg + 1) * 32 + r] = (__bf16)q.y;
      sBt[(cg + 2) * 32 + r] = (__bf16)q.z;
      sBt[(cg + 3) * 32 + r] = (__bf16)q.w;
    }
    __syncthreads();

    v8bf alo = *(const v8bf*)&sA[lrow * 32 + g * 8];
    v8bf ahi = *(const v8bf*)&sA[lrow * 32 + 16 + g * 8];
    v16bf a  = cat16(alo, ahi);
#pragma unroll
    for (int t = 0; t < 2; ++t) {
      v16bf bm = cat16(*(const v8bf*)&sBt[(wave * 32 + t * 16 + lrow) * 32 + g * 8],
                       *(const v8bf*)&sBt[(wave * 32 + t * 16 + lrow) * 32 + 16 + g * 8]);
      acc[t] = __builtin_amdgcn_wmma_f32_16x16x32_bf16(false, a, false, bm,
                                                       (short)0, acc[t],
                                                       false, false);
    }
    __syncthreads();
  }

  const int b  = win >> 6;
  const int wy = (win & 63) >> 3;
  const int wx = win & 7;
#pragma unroll
  for (int t = 0; t < 2; ++t) {
    const int col = c0 + wave * 32 + t * 16 + lrow;
    const float bias = proj_b[col];
#pragma unroll
    for (int i = 0; i < 8; ++i) {
      int m   = i + 8 * g;
      int tok = row0 + m;
      if (tok < NTOK) {
        int ty = tok / 7, tx = tok % 7;
        long L = (long)(wy * 7 + ty) * 56 + (wx * 7 + tx);
        out[((long)b * 3136 + L) * CDIM + col] = acc[t][i] + bias;
      }
    }
  }
}

// ---------------------------------------------------------------------------
extern "C" void kernel_launch(void* const* d_in, const int* in_sizes, int n_in,
                              void* d_out, int out_size, void* d_ws, size_t ws_size,
                              hipStream_t stream) {
  const float* hidden = (const float*)d_in[0];
  const float* qkv_w  = (const float*)d_in[1];
  const float* qkv_b  = (const float*)d_in[2];
  const float* proj_w = (const float*)d_in[3];
  const float* proj_b = (const float*)d_in[4];
  const float* relb   = (const float*)d_in[5];
  __bf16* ws  = (__bf16*)d_ws;
  float*  out = (float*)d_out;

  // 1) QKV: rows = 2048 windows * 4 row-tiles, cols = 1152 / 128
  swin_qkv_kernel<<<dim3(8192, 9), 128, 0, stream>>>(hidden, qkv_w, qkv_b, ws);
  // 2) attention: one block per (window, head)
  swin_attn_kernel<<<dim3(NWH), 128, 0, stream>>>(relb, ws);
  // 3) projection + window reverse: cols = 384 / 128
  swin_proj_kernel<<<dim3(8192, 3), 128, 0, stream>>>(proj_w, proj_b, ws, out);
}
